// CondConvDecoder_23407571763755
// MI455X (gfx1250) — compile-verified, matrix operand order checked by reading, not run on hardware
//
#include <hip/hip_runtime.h>

// ---------------------------------------------------------------------------
// CondConv 3D transposed-conv decoder for gfx1250 (MI455X).
//
// Conv layers = implicit GEMM on V_WMMA_F32_16X16X32_BF16 (bf16 in, f32 acc).
// Activations are pre-transformed per layer into channels-last bf16
// (xcl[b][spatial][CiPad], ci zero-padded, +1 all-zero spatial row) so the
// WMMA K-loop is pure b128 loads + wmma: invalid taps read the zero row,
// padded channels are zero, no masks/cvts in the hot loop. Strides are
// template parameters so tap indexing is shift/and only.
//
// d_in layout (setup_inputs dict order, nested dicts flattened depth-first):
//   0:x  1:routing_weights
//   2..13  resp1 {W1,b1,W2,b2,W3,b3,g1,be1,g2,be2,g3,be3}
//   14..25 res0   26..37 res1   38..49 res2
//   50:conv1_W 51:conv1_b 52:bn1_g 53:bn1_b 54:conv2_W 55:conv2_b
// ---------------------------------------------------------------------------

typedef __attribute__((ext_vector_type(16))) __bf16 v16bf;
typedef __attribute__((ext_vector_type(8)))  float  v8f;

union BFrag {
  v16bf v;
  unsigned short u[16];
  unsigned int   w[8];
};

__device__ __forceinline__ unsigned short f32_to_bf16_rne(float f) {
  unsigned int u = __builtin_bit_cast(unsigned int, f);
  u += 0x7fffu + ((u >> 16) & 1u);
  return (unsigned short)(u >> 16);
}

// -------------------- expert-mixing kernels (weights -> bf16) ---------------
// convT weights W[K][Ci][Co][27] -> wgt[b][tap][coPad][ciPad], taps unflipped.
__global__ void mix_wT_kernel(const float* __restrict__ W,
                              const float* __restrict__ rw,
                              unsigned short* __restrict__ wgt,
                              int Ci, int Co, int CiPad, int CoPad, int rwCol) {
  long i = (long)blockIdx.x * blockDim.x + threadIdx.x;
  long total = (long)8 * 27 * CoPad * CiPad;
  if (i >= total) return;
  int ci = (int)(i % CiPad); long r = i / CiPad;
  int co = (int)(r % CoPad); r /= CoPad;
  int tap = (int)(r % 27);   int b = (int)(r / 27);
  float v = 0.f;
  if (ci < Ci && co < Co) {
    const float* rwb = rw + ((long)b * 29 + rwCol) * 3;
#pragma unroll
    for (int k = 0; k < 3; ++k)
      v += rwb[k] * W[(((long)k * Ci + ci) * Co + co) * 27 + tap];
  }
  wgt[i] = f32_to_bf16_rne(v);
}

// plain conv weights W[K][Co][Ci][27] -> same layout, taps FLIPPED (26-tap),
// so a standard conv becomes convT(stride=1) in the shared conv kernel.
__global__ void mix_wC_kernel(const float* __restrict__ W,
                              const float* __restrict__ rw,
                              unsigned short* __restrict__ wgt,
                              int Ci, int Co, int CiPad, int CoPad, int rwCol) {
  long i = (long)blockIdx.x * blockDim.x + threadIdx.x;
  long total = (long)8 * 27 * CoPad * CiPad;
  if (i >= total) return;
  int ci = (int)(i % CiPad); long r = i / CiPad;
  int co = (int)(r % CoPad); r /= CoPad;
  int tap = (int)(r % 27);   int b = (int)(r / 27);
  float v = 0.f;
  if (ci < Ci && co < Co) {
    const float* rwb = rw + ((long)b * 29 + rwCol) * 3;
#pragma unroll
    for (int k = 0; k < 3; ++k)
      v += rwb[k] * W[(((long)k * Co + co) * Ci + ci) * 27 + (26 - tap)];
  }
  wgt[i] = f32_to_bf16_rne(v);
}

__global__ void mix_bias_kernel(const float* __restrict__ Bb,
                                const float* __restrict__ rw,
                                float* __restrict__ bias,
                                int Co, int CoPad, int rwCol) {
  int i = blockIdx.x * blockDim.x + threadIdx.x;
  if (i >= 8 * CoPad) return;
  int co = i % CoPad, b = i / CoPad;
  float v = 0.f;
  if (co < Co) {
    const float* rwb = rw + ((long)b * 29 + rwCol) * 3;
#pragma unroll
    for (int k = 0; k < 3; ++k) v += rwb[k] * Bb[k * Co + co];
  }
  bias[i] = v;
}

// ------------- activation transform: NCDHW f32 -> channels-last bf16 --------
// xcl[b][s][CiPad] for s in [0,Sin]; row s==Sin and ci>=Ci are zero.
// LDS-tiled transpose: 32 channels x 64 spatial per block.
// grid: (ceil((Sin+1)/64), CiPad/32, 8)
__global__ void __launch_bounds__(256)
to_cl_bf16_kernel(const float* __restrict__ x, unsigned int* __restrict__ xcl,
                  int Ci, int CiPad, int Sin) {
  const int b  = blockIdx.z;
  const int c0 = blockIdx.y * 32;
  const int s0 = blockIdx.x * 64;
  __shared__ float tile[32][65];

  // coalesced read: consecutive threads -> consecutive spatial s
  const int ts = threadIdx.x & 63;
  const int tr = threadIdx.x >> 6;               // 4 channel rows per pass
  const int s  = s0 + ts;
#pragma unroll
  for (int r = tr; r < 32; r += 4) {
    const int ci = c0 + r;
    float v = 0.f;
    if (ci < Ci && s < Sin) v = x[((long)b * Ci + ci) * Sin + s];
    tile[r][ts] = v;
  }
  __syncthreads();

  // coalesced write: consecutive threads -> consecutive channel dwords
  const int cph = CiPad >> 1;
  const int cp  = threadIdx.x & 15;              // 16 dwords = 32 channels
  const int sr  = threadIdx.x >> 4;              // 16 spatial rows per pass
  unsigned int* ob = xcl + (long)b * (Sin + 1) * cph + (c0 >> 1);
#pragma unroll
  for (int r = sr; r < 64; r += 16) {
    const int so = s0 + r;
    if (so <= Sin) {
      const unsigned int lo = f32_to_bf16_rne(tile[cp * 2][r]);
      const unsigned int hi = f32_to_bf16_rne(tile[cp * 2 + 1][r]);
      ob[(long)so * cph + cp] = lo | (hi << 16);
    }
  }
}

// -------------------- implicit-GEMM conv_transpose via WMMA -----------------
// Gather form: out[b,co,od,oh,ow] = bias + sum_{tap,ci} xcl[b,soff,ci] *
// wgt[b,tap,co,ci], soff = spatial(id,ih,iw) or the zero row when the tap is
// invalid. One wave computes a 16(co) x 32(pos) tile: one A fragment feeds two
// accumulators (two position tiles) -> 2 WMMA per weight fetch.
template <int SD, int SH, int SW>
__global__ void __launch_bounds__(256)
condconv_wmma_kernel(const unsigned short* __restrict__ xcl,
                     const unsigned short* __restrict__ wgt,
                     const float* __restrict__ bias,
                     float* __restrict__ out,
                     int Co, int CiPad, int CoPad,
                     int Di, int Hi, int Wi, int Do, int Ho, int Wo) {
  const int lane = threadIdx.x & 31;
  const int wv   = threadIdx.x >> 5;
  const int b    = blockIdx.z;
  const int coBase = blockIdx.y * 16;
  const int Npos = Do * Ho * Wo;
  const int posBase = (blockIdx.x * 8 + wv) * 32;
  if (posBase >= Npos) return;               // uniform per wave

  const int m  = lane & 15;                  // A row (co); also B col (pos)
  const int hf = lane >> 4;                  // lane half -> K-group select
  const int Sin = Di * Hi * Wi;              // zero row index

  // two position tiles
  const int p0 = posBase + m;
  const int p1 = posBase + 16 + m;
  const bool pv0 = (p0 < Npos), pv1 = (p1 < Npos);
  const int pc0 = pv0 ? p0 : 0, pc1 = pv1 ? p1 : 0;
  const int ow0 = pc0 % Wo, t00 = pc0 / Wo, oh0 = t00 % Ho, od0 = t00 / Ho;
  const int ow1 = pc1 % Wo, t01 = pc1 / Wo, oh1 = t01 % Ho, od1 = t01 / Ho;

  const unsigned short* xb = xcl + (long)b * (Sin + 1) * CiPad;
  const unsigned short* wb = wgt + (long)b * 27 * CoPad * CiPad;

  v8f acc0 = {}, acc1 = {};
  const int nkb = CiPad >> 5;

  for (int tap = 0; tap < 27; ++tap) {
    const int kd = tap / 9, kh = (tap / 3) % 3, kw = tap % 3;

    // tile 0 validity / source row (compile-time strides: &, >> only)
    const int dn0 = od0 + 1 - kd, hn0 = oh0 + 1 - kh, wn0 = ow0 + 1 - kw;
    bool v0 = pv0 && dn0 >= 0 && hn0 >= 0 && wn0 >= 0 &&
              (dn0 % SD) == 0 && (hn0 % SH) == 0 && (wn0 % SW) == 0;
    const int id0 = dn0 / SD, ih0 = hn0 / SH, iw0 = wn0 / SW;
    v0 = v0 && id0 < Di && ih0 < Hi && iw0 < Wi;
    const int soff0 = v0 ? ((id0 * Hi + ih0) * Wi + iw0) : Sin;

    const int dn1 = od1 + 1 - kd, hn1 = oh1 + 1 - kh, wn1 = ow1 + 1 - kw;
    bool v1 = pv1 && dn1 >= 0 && hn1 >= 0 && wn1 >= 0 &&
              (dn1 % SD) == 0 && (hn1 % SH) == 0 && (wn1 % SW) == 0;
    const int id1 = dn1 / SD, ih1 = hn1 / SH, iw1 = wn1 / SW;
    v1 = v1 && id1 < Di && ih1 < Hi && iw1 < Wi;
    const int soff1 = v1 ? ((id1 * Hi + ih1) * Wi + iw1) : Sin;

    const unsigned short* wrow = wb + ((long)tap * CoPad + (coBase + m)) * CiPad;
    const unsigned short* brow0 = xb + (long)soff0 * CiPad;
    const unsigned short* brow1 = xb + (long)soff1 * CiPad;

    for (int kb = 0; kb < nkb; ++kb) {
      const int kbase = kb << 5;
      BFrag af, bf0, bf1;
      const unsigned int* aw = (const unsigned int*)(wrow + kbase);
      const unsigned int* b0 = (const unsigned int*)(brow0 + kbase);
      const unsigned int* b1 = (const unsigned int*)(brow1 + kbase);
#pragma unroll
      for (int j = 0; j < 8; ++j) {
        // 16-bit 16x32 fragment layout (ISA 7.12.2): pairs per dword,
        // j<4 -> dwords {4h..4h+3}, j>=4 -> dwords {8+4h..11+4h}: 2x b128 each.
        const int dw = (j < 4) ? (j + 4 * hf) : (8 + (j - 4) + 4 * hf);
        af.w[j]  = aw[dw];
        bf0.w[j] = b0[dw];
        bf1.w[j] = b1[dw];
      }
      acc0 = __builtin_amdgcn_wmma_f32_16x16x32_bf16(
          false, af.v, false, bf0.v, (short)0, acc0, false, false);
      acc1 = __builtin_amdgcn_wmma_f32_16x16x32_bf16(
          false, af.v, false, bf1.v, (short)0, acc1, false, false);
    }
  }

  float* ob = out + (long)b * Co * Npos;
  const float* bb = bias + b * CoPad;
#pragma unroll
  for (int j = 0; j < 8; ++j) {              // C/D layout: m = j + 8*half
    const int co = coBase + j + 8 * hf;
    const bool cv = (co < Co);
    if (cv && pv0) ob[(long)co * Npos + p0] = acc0[j] + bb[co];
    if (cv && pv1) ob[(long)co * Npos + p1] = acc1[j] + bb[co];
  }
}

// ---------------- BatchNorm (train-mode batch stats), two-stage -------------
// Stage 1: fixed spatial chunking -> per-(channel, chunk) partial {sum, sum2}.
// grid: (nch, C). Deterministic (fixed partition + fixed reduction tree).
__global__ void __launch_bounds__(256)
bn_stats_part_kernel(const float* __restrict__ t, float* __restrict__ part,
                     int C, int S) {
  const int c = blockIdx.y, ch = blockIdx.x, nch = gridDim.x;
  const int per = (S + nch - 1) / nch;
  const int sBeg = ch * per;
  const int sEnd = (sBeg + per < S) ? (sBeg + per) : S;
  float s = 0.f, s2 = 0.f;
  for (int b = 0; b < 8; ++b) {
    const float* tb = t + ((long)b * C + c) * S;
    for (int sp = sBeg + threadIdx.x; sp < sEnd; sp += 256) {
      const float v = tb[sp];
      s += v; s2 += v * v;
    }
  }
  __shared__ float sh[256], sh2[256];
  sh[threadIdx.x] = s; sh2[threadIdx.x] = s2;
  __syncthreads();
  for (int o = 128; o > 0; o >>= 1) {
    if ((int)threadIdx.x < o) {
      sh[threadIdx.x]  += sh[threadIdx.x + o];
      sh2[threadIdx.x] += sh2[threadIdx.x + o];
    }
    __syncthreads();
  }
  if (threadIdx.x == 0) {
    part[((long)c * nch + ch) * 2]     = sh[0];
    part[((long)c * nch + ch) * 2 + 1] = sh2[0];
  }
}

// Stage 2: fold partials, emit per-channel {scale, shift}. grid: (C)
__global__ void __launch_bounds__(256)
bn_stats_final_kernel(const float* __restrict__ part,
                      const float* __restrict__ g, const float* __restrict__ be,
                      float* __restrict__ stats, int nch, int S) {
  const int c = blockIdx.x;
  float s = 0.f, s2 = 0.f;
  for (int i = threadIdx.x; i < nch; i += 256) {
    s  += part[((long)c * nch + i) * 2];
    s2 += part[((long)c * nch + i) * 2 + 1];
  }
  __shared__ float sh[256], sh2[256];
  sh[threadIdx.x] = s; sh2[threadIdx.x] = s2;
  __syncthreads();
  for (int o = 128; o > 0; o >>= 1) {
    if ((int)threadIdx.x < o) {
      sh[threadIdx.x]  += sh[threadIdx.x + o];
      sh2[threadIdx.x] += sh2[threadIdx.x + o];
    }
    __syncthreads();
  }
  if (threadIdx.x == 0) {
    const float count = (float)((long)8 * S);
    const float mean = sh[0] / count;
    const float var  = sh2[0] / count - mean * mean;
    const float sc   = g[c] * rsqrtf(var + 1e-5f);
    stats[2 * c]     = sc;
    stats[2 * c + 1] = be[c] - mean * sc;
  }
}

// grid: (ceil(S/256), C, 8)
__global__ void bn_lrelu_kernel(float* __restrict__ t,
                                const float* __restrict__ stats, int C, int S) {
  const int sp = blockIdx.x * blockDim.x + threadIdx.x;
  if (sp >= S) return;
  const int c = blockIdx.y, b = blockIdx.z;
  const long i = ((long)b * C + c) * S + sp;
  const float v = t[i] * stats[2 * c] + stats[2 * c + 1];
  t[i] = v >= 0.f ? v : 0.2f * v;
}

// a = lrelu(bn(a; sa)) + lrelu(bn(b; sb))   (block residual merge, in place)
__global__ void bn_lrelu_add_kernel(float* __restrict__ a,
                                    const float* __restrict__ bsrc,
                                    const float* __restrict__ sa,
                                    const float* __restrict__ sb,
                                    int C, int S) {
  const int sp = blockIdx.x * blockDim.x + threadIdx.x;
  if (sp >= S) return;
  const int c = blockIdx.y, b = blockIdx.z;
  const long i = ((long)b * C + c) * S + sp;
  const float v1 = a[i]    * sa[2 * c] + sa[2 * c + 1];
  const float v2 = bsrc[i] * sb[2 * c] + sb[2 * c + 1];
  a[i] = (v1 >= 0.f ? v1 : 0.2f * v1) + (v2 >= 0.f ? v2 : 0.2f * v2);
}

// -------------------- host-side orchestration -------------------------------
static inline int ru16(int v) { return (v + 15) & ~15; }
static inline int ru32(int v) { return (v + 31) & ~31; }

// workspace offsets (bytes)
static const size_t O_WGT  = 0;                                  // 28.3 MB (bf16 max)
static const size_t O_BIAS = O_WGT  + (size_t)8 * 27 * 256 * 256 * 2;
static const size_t O_SA   = O_BIAS + (size_t)8 * 256 * 4;
static const size_t O_SB   = O_SA   + 512 * 4;
static const size_t O_PART = O_SB   + 512 * 4;                   // 512 KB partials
static const size_t O_T1   = O_PART + (size_t)256 * 256 * 2 * 4; // 226.5 MB cap
static const size_t O_Q0   = O_T1 + (size_t)8 * 32 * 221184 * 4; // 113.2 MB
static const size_t O_Q1   = O_Q0 + (size_t)8 * 16 * 221184 * 4;
static const size_t O_T3   = O_Q1 + (size_t)8 * 16 * 221184 * 4;
static const size_t O_XCL  = O_T3 + (size_t)8 * 16 * 221184 * 4; // 113.2 MB cap

static void launch_stats(const float* t, const float* g, const float* be,
                         float* part, float* stats, int C, int S,
                         hipStream_t stream) {
  int nch = (int)(((long)8 * S) / 16384);
  if (nch < 1) nch = 1;
  if (nch > 256) nch = 256;
  bn_stats_part_kernel<<<dim3(nch, C), 256, 0, stream>>>(t, part, C, S);
  bn_stats_final_kernel<<<dim3(C), 256, 0, stream>>>(part, g, be, stats, nch, S);
}

static void launch_cond_layer(const float* xin, const float* W, const float* Bb,
                              const float* rw, int rwCol, bool isConvT,
                              int Ci, int Co,
                              int Di, int Hi, int Wi, int Do, int Ho, int Wo,
                              int sd, int sh, int sw,
                              unsigned short* wgt, float* bias,
                              unsigned short* xcl, float* out,
                              hipStream_t stream) {
  const int CiPad = ru32(Ci), CoPad = ru16(Co);
  const long wtot = (long)8 * 27 * CoPad * CiPad;
  const int thr = 256;
  if (isConvT)
    mix_wT_kernel<<<dim3((unsigned)((wtot + thr - 1) / thr)), thr, 0, stream>>>(
        W, rw, wgt, Ci, Co, CiPad, CoPad, rwCol);
  else
    mix_wC_kernel<<<dim3((unsigned)((wtot + thr - 1) / thr)), thr, 0, stream>>>(
        W, rw, wgt, Ci, Co, CiPad, CoPad, rwCol);
  mix_bias_kernel<<<dim3((8 * CoPad + 255) / 256), 256, 0, stream>>>(
      Bb, rw, bias, Co, CoPad, rwCol);

  const int Sin = Di * Hi * Wi;
  dim3 tg((Sin + 1 + 63) / 64, CiPad / 32, 8);
  to_cl_bf16_kernel<<<tg, 256, 0, stream>>>(xin, (unsigned int*)xcl, Ci, CiPad, Sin);

  const int Npos = Do * Ho * Wo;
  dim3 g((Npos + 255) / 256, CoPad / 16, 8);
  if (sd == 2 && sh == 2 && sw == 1)
    condconv_wmma_kernel<2, 2, 1><<<g, 256, 0, stream>>>(
        xcl, wgt, bias, out, Co, CiPad, CoPad, Di, Hi, Wi, Do, Ho, Wo);
  else if (sd == 2 && sh == 2 && sw == 2)
    condconv_wmma_kernel<2, 2, 2><<<g, 256, 0, stream>>>(
        xcl, wgt, bias, out, Co, CiPad, CoPad, Di, Hi, Wi, Do, Ho, Wo);
  else
    condconv_wmma_kernel<1, 1, 1><<<g, 256, 0, stream>>>(
        xcl, wgt, bias, out, Co, CiPad, CoPad, Di, Hi, Wi, Do, Ho, Wo);
}

struct BlkCfg {
  int base, fin, fout;
  int Di, Hi, Wi, Do, Ho, Wo, sd, sh, sw;
  int rw0;
};

extern "C" void kernel_launch(void* const* d_in, const int* in_sizes, int n_in,
                              void* d_out, int out_size, void* d_ws, size_t ws_size,
                              hipStream_t stream) {
  (void)in_sizes; (void)n_in; (void)out_size; (void)ws_size;
  const float* x  = (const float*)d_in[0];
  const float* rw = (const float*)d_in[1];

  char* ws = (char*)d_ws;
  unsigned short* WGT = (unsigned short*)(ws + O_WGT);
  float* BIAS = (float*)(ws + O_BIAS);
  float* SA   = (float*)(ws + O_SA);
  float* SB   = (float*)(ws + O_SB);
  float* PART = (float*)(ws + O_PART);
  float* T1   = (float*)(ws + O_T1);
  float* Q[2] = { (float*)(ws + O_Q0), (float*)(ws + O_Q1) };
  float* T3   = (float*)(ws + O_T3);
  unsigned short* XCL = (unsigned short*)(ws + O_XCL);

  static const BlkCfg blk[4] = {
      {  2, 256, 128,  6,  6,  3, 12, 12,  3, 2, 2, 1, 15 },  // resp1
      { 14, 128,  64, 12, 12,  3, 24, 24,  6, 2, 2, 2, 18 },  // res0
      { 26,  64,  32, 24, 24,  6, 48, 48, 12, 2, 2, 2, 21 },  // res1
      { 38,  32,  16, 48, 48, 12, 96, 96, 24, 2, 2, 2, 24 },  // res2
  };

  const float* blkIn = x;
  for (int i = 0; i < 4; ++i) {
    const BlkCfg& c = blk[i];
    const float* W1  = (const float*)d_in[c.base + 0];
    const float* b1  = (const float*)d_in[c.base + 1];
    const float* W2  = (const float*)d_in[c.base + 2];
    const float* b2  = (const float*)d_in[c.base + 3];
    const float* W3  = (const float*)d_in[c.base + 4];
    const float* b3  = (const float*)d_in[c.base + 5];
    const float* g1  = (const float*)d_in[c.base + 6];
    const float* be1 = (const float*)d_in[c.base + 7];
    const float* g2  = (const float*)d_in[c.base + 8];
    const float* be2 = (const float*)d_in[c.base + 9];
    const float* g3  = (const float*)d_in[c.base + 10];
    const float* be3 = (const float*)d_in[c.base + 11];
    const int Sout = c.Do * c.Ho * c.Wo;
    float* t1 = T1;
    float* t2 = Q[i & 1];
    dim3 eg((Sout + 255) / 256, 0, 8);

    // h1 = lrelu(bn(convT1(x)))          (fin -> fin, strided upsample)
    launch_cond_layer(blkIn, W1, b1, rw, c.rw0, true, c.fin, c.fin,
                      c.Di, c.Hi, c.Wi, c.Do, c.Ho, c.Wo, c.sd, c.sh, c.sw,
                      WGT, BIAS, XCL, t1, stream);
    launch_stats(t1, g1, be1, PART, SA, c.fin, Sout, stream);
    eg.y = c.fin;
    bn_lrelu_kernel<<<eg, 256, 0, stream>>>(t1, SA, c.fin, Sout);

    // t2 = convT2(h1)                    (fin -> fout, stride 1)
    launch_cond_layer(t1, W2, b2, rw, c.rw0 + 1, true, c.fin, c.fout,
                      c.Do, c.Ho, c.Wo, c.Do, c.Ho, c.Wo, 1, 1, 1,
                      WGT, BIAS, XCL, t2, stream);
    // t3 = convT3(x)                     (fin -> fout, strided shortcut)
    launch_cond_layer(blkIn, W3, b3, rw, c.rw0 + 2, true, c.fin, c.fout,
                      c.Di, c.Hi, c.Wi, c.Do, c.Ho, c.Wo, c.sd, c.sh, c.sw,
                      WGT, BIAS, XCL, T3, stream);
    launch_stats(t2, g2, be2, PART, SA, c.fout, Sout, stream);
    launch_stats(T3, g3, be3, PART, SB, c.fout, Sout, stream);
    eg.y = c.fout;
    bn_lrelu_add_kernel<<<eg, 256, 0, stream>>>(t2, T3, SA, SB, c.fout, Sout);

    blkIn = t2;
  }

  // conv1 (standard conv, 16 -> 8) + bn + lrelu
  const int Sf = 96 * 96 * 24;
  launch_cond_layer(blkIn, (const float*)d_in[50], (const float*)d_in[51],
                    rw, 27, false, 16, 8, 96, 96, 24, 96, 96, 24, 1, 1, 1,
                    WGT, BIAS, XCL, T1, stream);
  launch_stats(T1, (const float*)d_in[52], (const float*)d_in[53],
               PART, SA, 8, Sf, stream);
  {
    dim3 eg((Sf + 255) / 256, 8, 8);
    bn_lrelu_kernel<<<eg, 256, 0, stream>>>(T1, SA, 8, Sf);
  }
  // conv2 (standard conv, 8 -> 4) -> d_out
  launch_cond_layer(T1, (const float*)d_in[54], (const float*)d_in[55],
                    rw, 28, false, 8, 4, 96, 96, 24, 96, 96, 24, 1, 1, 1,
                    WGT, BIAS, XCL, (float*)d_out, stream);
}
